// MultiSimilarityLoss_6236292514414
// MI455X (gfx1250) — compile-verified, matrix operand order checked by reading, not run on hardware
//
#include <hip/hip_runtime.h>
#include <hip/hip_bf16.h>

typedef __attribute__((ext_vector_type(16))) _Float16 v16h;
typedef __attribute__((ext_vector_type(8)))  _Float16 v8h;
typedef __attribute__((ext_vector_type(8)))  float    v8f;

#define MS_ALPHA  2.0f
#define MS_BETA   50.0f
#define MS_BASE   0.5f
#define MS_MARGIN 0.1f
#define B_N 8192
#define D_K 256
#define WAVES_PER_BLOCK 8
#define COL_CHUNKS 16
#define TILES_PER_CHUNK ((B_N / 16) / COL_CHUNKS)   // 32 tiles of 16 columns each

#define POS_INF  __int_as_float(0x7F800000)
#define NEG_INF  __int_as_float(0xFF800000)

// ---- order-preserving float <-> uint mapping for atomic min/max ----
__device__ __forceinline__ unsigned f2key(float f) {
    unsigned u = __float_as_uint(f);
    return (u & 0x80000000u) ? ~u : (u | 0x80000000u);
}
__device__ __forceinline__ float key2f(unsigned k) {
    unsigned u = (k & 0x80000000u) ? (k & 0x7FFFFFFFu) : ~k;
    return __uint_as_float(u);
}

// ---- WMMA fragment gather ----
// 16-bit A (16x32 MxK) and B (32x16 KxN) fragments share the same per-lane
// gather on row-major [row][k] fp16 data:
//   lane L: row/col = base + (L & 15), koff = (L >= 16) ? 8 : 0
//   halves 0..7  = k in [kc*32 + koff,      kc*32 + koff + 7]
//   halves 8..15 = k in [kc*32 + koff + 16, kc*32 + koff + 23]
__device__ __forceinline__ v16h frag_from(const _Float16* p) {
    union { v16h v; v8h h[2]; } u;
    u.h[0] = *(const v8h*)(p);
    u.h[1] = *(const v8h*)(p + 16);
    return u.v;
}

// ---- fp32 -> fp16 conversion of embeddings ----
__global__ void ms_convert(const float* __restrict__ emb, _Float16* __restrict__ e16, int n) {
    int i = blockIdx.x * blockDim.x + threadIdx.x;
    if (i < n) e16[i] = (_Float16)emb[i];
}

// ---- workspace init (harness poisons ws; must re-init every launch) ----
__global__ void ms_init(unsigned* __restrict__ pminKey, unsigned* __restrict__ nmaxKey,
                        float* __restrict__ sumPos, float* __restrict__ sumNeg,
                        float* __restrict__ accum) {
    int i = blockIdx.x * blockDim.x + threadIdx.x;
    if (i < B_N) {
        pminKey[i] = 0xFFFFFFFFu;  // above key(+inf): min sentinel
        nmaxKey[i] = 0x00000000u;  // below key(-inf): max sentinel
        sumPos[i]  = 0.0f;
        sumNeg[i]  = 0.0f;
    }
    if (i == 0) { accum[0] = 0.0f; accum[1] = 0.0f; }
}

// ============================================================================
// Fused GEMM+elementwise passes.
// Block = 8 waves; wave w owns row strip (blockIdx.x*8 + w) of 16 rows.
// All 8 waves share one column chunk: each 8KB B-tile (16 cols x 256 k, a
// contiguous slab of row-major fp16) is staged into LDS once per block,
// double-buffered, then fragments come from ds_load_b128.
// PASS: 1 = pos_min/neg_max reductions, 2 = hard-mask exp sums.
// ============================================================================
template <int PASS>
__device__ __forceinline__ void ms_pass_body(const _Float16* __restrict__ e16,
                                             const int* __restrict__ labels,
                                             unsigned* __restrict__ pminKey,
                                             unsigned* __restrict__ nmaxKey,
                                             float* __restrict__ sumPos,
                                             float* __restrict__ sumNeg) {
    __shared__ __align__(16) _Float16 lds_tile[2][16 * D_K];   // 2 x 8KB

    const unsigned tid  = threadIdx.x;
    const unsigned lane = tid & 31u;
    const int      w    = (int)(tid >> 5);
    const int row_base  = ((int)blockIdx.x * WAVES_PER_BLOCK + w) * 16;
    const int chunk     = (int)blockIdx.y;
    const int col0      = chunk * TILES_PER_CHUNK * 16;
    const int hi        = (int)(lane >> 4) & 1;
    const int n         = (int)(lane & 15u);
    const int koff      = hi ? 8 : 0;

    // ---- per-wave A fragments (held in registers across the column sweep) ----
    v16h a[8];
#pragma unroll
    for (int kc = 0; kc < 8; ++kc)
        a[kc] = frag_from(e16 + (size_t)(row_base + n) * D_K + kc * 32 + koff);

    int lrow[8];
    float pmr[8], nmr[8];
#pragma unroll
    for (int r = 0; r < 8; ++r) {
        int rowg = row_base + r + hi * 8;
        lrow[r]  = labels[rowg];
        if (PASS == 2) {
            pmr[r] = key2f(pminKey[rowg]);  // untouched sentinel -> NaN -> empty mask (matches ref)
            nmr[r] = key2f(nmaxKey[rowg]);
        }
    }

    float acc0[8], acc1[8];   // pass1: pos_min/neg_max, pass2: sum_pos/sum_neg
#pragma unroll
    for (int r = 0; r < 8; ++r) {
        acc0[r] = (PASS == 1) ? POS_INF : 0.0f;
        acc1[r] = (PASS == 1) ? NEG_INF : 0.0f;
    }

    // ---- stage tile 0 into LDS buffer 0 (contiguous 8KB memcpy) ----
    const uint4* __restrict__ gch = (const uint4*)(e16 + (size_t)col0 * D_K);
    {
        uint4* l = (uint4*)&lds_tile[0][0];
        l[tid * 2]     = gch[tid * 2];
        l[tid * 2 + 1] = gch[tid * 2 + 1];
    }
    __syncthreads();

    const int STRIDE = 16 * D_K / 8;  // uint4 elements per tile = 512

    for (int t = 0; t < TILES_PER_CHUNK; ++t) {
        const int b        = t & 1;
        const int col_base = col0 + t * 16;

        // issue next tile's global loads early; LDS store happens after compute
        uint4 r0, r1;
        if (t + 1 < TILES_PER_CHUNK) {
            const uint4* g = gch + (size_t)(t + 1) * STRIDE;
            r0 = g[tid * 2];
            r1 = g[tid * 2 + 1];
        }
        if (t + 2 < TILES_PER_CHUNK)
            __builtin_prefetch(gch + (size_t)(t + 2) * STRIDE, 0, 0);

        // ---- 16x16 sim tile: 8 accumulate-chained WMMAs fed from LDS ----
        v8f c = {};
        const _Float16* lbase = &lds_tile[b][(size_t)n * D_K + koff];
#pragma unroll
        for (int kc = 0; kc < 8; ++kc) {
            v16h bf = frag_from(lbase + kc * 32);
            c = __builtin_amdgcn_wmma_f32_16x16x32_f16(false, a[kc], false, bf,
                                                       (short)0, c, false, false);
        }

        // ---- branchless elementwise (v_cndmask + v_min/max/add, no EXEC churn) ----
        const int colg = col_base + n;
        const int lc   = labels[colg];
#pragma unroll
        for (int r = 0; r < 8; ++r) {
            const int   rowg = row_base + r + hi * 8;
            const float s    = c[r];
            const bool  same = (lrow[r] == lc);
            if (PASS == 1) {
                const bool pos = same && (rowg != colg);
                acc0[r] = fminf(acc0[r], pos  ? s : POS_INF);
                acc1[r] = fmaxf(acc1[r], same ? NEG_INF : s);
            } else {
                const bool hp = same && (rowg != colg) && (s - MS_MARGIN < nmr[r]);
                const bool hn = (!same) && (s + MS_MARGIN > pmr[r]);
                // scores bounded in [-75,25]: no max-shift needed, fp32-safe
                acc0[r] += hp ? __expf(-MS_ALPHA * (s - MS_BASE)) : 0.0f;
                acc1[r] += hn ? __expf( MS_BETA  * (s - MS_BASE)) : 0.0f;
            }
        }

        // ---- commit next tile to the other LDS buffer ----
        if (t + 1 < TILES_PER_CHUNK) {
            uint4* l = (uint4*)&lds_tile[b ^ 1][0];
            l[tid * 2]     = r0;
            l[tid * 2 + 1] = r1;
        }
        __syncthreads();
    }

    // ---- reduce across the 16-lane group (xor stays inside each half-wave) ----
#pragma unroll
    for (int m = 1; m < 16; m <<= 1) {
#pragma unroll
        for (int r = 0; r < 8; ++r) {
            if (PASS == 1) {
                acc0[r] = fminf(acc0[r], __shfl_xor(acc0[r], m, 32));
                acc1[r] = fmaxf(acc1[r], __shfl_xor(acc1[r], m, 32));
            } else {
                acc0[r] += __shfl_xor(acc0[r], m, 32);
                acc1[r] += __shfl_xor(acc1[r], m, 32);
            }
        }
    }
    if (n == 0) {
#pragma unroll
        for (int r = 0; r < 8; ++r) {
            int rowg = row_base + r + hi * 8;
            if (PASS == 1) {
                atomicMin(&pminKey[rowg], f2key(acc0[r]));
                atomicMax(&nmaxKey[rowg], f2key(acc1[r]));
            } else {
                atomicAdd(&sumPos[rowg], acc0[r]);
                atomicAdd(&sumNeg[rowg], acc1[r]);
            }
        }
    }
}

__global__ __launch_bounds__(256) void ms_pass1(const _Float16* __restrict__ e16,
                                                const int* __restrict__ labels,
                                                unsigned* __restrict__ pminKey,
                                                unsigned* __restrict__ nmaxKey) {
    ms_pass_body<1>(e16, labels, pminKey, nmaxKey, nullptr, nullptr);
}

__global__ __launch_bounds__(256) void ms_pass2(const _Float16* __restrict__ e16,
                                                const int* __restrict__ labels,
                                                unsigned* __restrict__ pminKey,
                                                unsigned* __restrict__ nmaxKey,
                                                float* __restrict__ sumPos,
                                                float* __restrict__ sumNeg) {
    ms_pass_body<2>(e16, labels, pminKey, nmaxKey, sumPos, sumNeg);
}

// ---- finalize: per-row terms, validity, global accumulate ----
__global__ __launch_bounds__(256) void ms_finalize(const float* __restrict__ sumPos,
                                                   const float* __restrict__ sumNeg,
                                                   float* __restrict__ accum) {
    __shared__ float sl[256];
    __shared__ float sv[256];
    int row = blockIdx.x * blockDim.x + threadIdx.x;
    float l = 0.0f, v = 0.0f;
    if (row < B_N) {
        float sp = sumPos[row], sn = sumNeg[row];
        if (sp > 0.0f && sn > 0.0f) {   // any(hard_*): exp terms strictly positive
            v = 1.0f;
            l = log1pf(sp) / MS_ALPHA + log1pf(sn) / MS_BETA;
        }
    }
    sl[threadIdx.x] = l;
    sv[threadIdx.x] = v;
    __syncthreads();
    for (int s = 128; s > 0; s >>= 1) {
        if ((int)threadIdx.x < s) {
            sl[threadIdx.x] += sl[threadIdx.x + s];
            sv[threadIdx.x] += sv[threadIdx.x + s];
        }
        __syncthreads();
    }
    if (threadIdx.x == 0) {
        atomicAdd(&accum[0], sl[0]);
        atomicAdd(&accum[1], sv[0]);
    }
}

__global__ void ms_div(const float* __restrict__ accum, float* __restrict__ out) {
    out[0] = accum[0] / fmaxf(accum[1], 1.0f);
}

extern "C" void kernel_launch(void* const* d_in, const int* in_sizes, int n_in,
                              void* d_out, int out_size, void* d_ws, size_t ws_size,
                              hipStream_t stream) {
    const float* emb    = (const float*)d_in[0];
    const int*   labels = (const int*)d_in[1];
    float*       out    = (float*)d_out;

    // workspace layout: fp16 embeddings (4 MiB) + per-row reductions + accumulators
    char*      ws      = (char*)d_ws;
    _Float16*  e16     = (_Float16*)ws;
    unsigned*  pminKey = (unsigned*)(ws + (size_t)B_N * D_K * 2);
    unsigned*  nmaxKey = pminKey + B_N;
    float*     sumPos  = (float*)(nmaxKey + B_N);
    float*     sumNeg  = sumPos + B_N;
    float*     accum   = sumNeg + B_N;

    const int nelem = B_N * D_K;
    ms_convert<<<(nelem + 255) / 256, 256, 0, stream>>>(emb, e16, nelem);
    ms_init<<<(B_N + 255) / 256, 256, 0, stream>>>(pminKey, nmaxKey, sumPos, sumNeg, accum);

    dim3 grid(B_N / (16 * WAVES_PER_BLOCK), COL_CHUNKS);   // 64 x 16 blocks of 8 waves
    ms_pass1<<<grid, 32 * WAVES_PER_BLOCK, 0, stream>>>(e16, labels, pminKey, nmaxKey);
    ms_pass2<<<grid, 32 * WAVES_PER_BLOCK, 0, stream>>>(e16, labels, pminKey, nmaxKey,
                                                        sumPos, sumNeg);

    ms_finalize<<<B_N / 256, 256, 0, stream>>>(sumPos, sumNeg, accum);
    ms_div<<<1, 1, 0, stream>>>(accum, out);
}
